// InfoNCELoss_43757126812327
// MI455X (gfx1250) — matrix operand order repaired and checked by reading, not compiled, not run on hardware
//
#include <hip/hip_runtime.h>
#include <hip/hip_bf16.h>

typedef __attribute__((ext_vector_type(16))) _Float16 v16h;
typedef __attribute__((ext_vector_type(8)))  _Float16 v8h;
typedef __attribute__((ext_vector_type(8)))  float    v8f;

#define B_   32
#define C_   256
#define HF_  64
#define WF_  64
#define N_   256          // 16x16 grid points
#define HO_  512
#define INV_T 14.285714285714286f   // 1/0.07

// ---------------------------------------------------------------------------
// Kernel 1: gather 256 grid points per (batch, map), L2-normalize over C=256,
// store as f16 rows [b][n][c]. One wave per (map,b,n): 8 channels per lane.
// ---------------------------------------------------------------------------
__global__ void gather_norm(const float* __restrict__ fdepth,
                            const float* __restrict__ frgb,
                            _Float16* __restrict__ dOut,
                            _Float16* __restrict__ rOut) {
  const int task = blockIdx.x * (blockDim.x >> 5) + (threadIdx.x >> 5); // [0, 2*B*N)
  const int lane = threadIdx.x & 31;
  const int which = task >> 13;          // 8192 = B_*N_
  const int pn    = task & 8191;
  const int b     = pn >> 8;
  const int n     = pn & 255;
  const int y     = (n >> 4) << 2;       // 4*(n/16)
  const int x     = (n & 15) << 2;       // 4*(n%16)

  const float* src = which ? frgb : fdepth;
  _Float16*    dst = which ? rOut : dOut;

  const size_t base = (size_t)b * C_ * HF_ * WF_ + (size_t)y * WF_ + x;

  float vals[8];
  float ss = 0.f;
#pragma unroll
  for (int i = 0; i < 8; ++i) {
    const int c = lane + (i << 5);
    const float v = src[base + (size_t)c * (HF_ * WF_)];
    vals[i] = v;
    ss += v * v;
  }
#pragma unroll
  for (int m = 16; m > 0; m >>= 1) ss += __shfl_xor(ss, m, 32);

  const float scale = 1.0f / fmaxf(sqrtf(ss), 1e-12f);

  _Float16* d = dst + (((size_t)b << 8) + n) * C_;
#pragma unroll
  for (int i = 0; i < 8; ++i) {
    const int c = lane + (i << 5);
    d[c] = (_Float16)(vals[i] * scale);
  }
}

// ---------------------------------------------------------------------------
// Kernel 2: grid validity flags. vm_f[b,0,4*(n/16),4*(n%16)] with ds=8
// => full-res pixel (32*(n/16), 32*(n%16)).
// ---------------------------------------------------------------------------
__global__ void compute_valid(const float* __restrict__ vm, int* __restrict__ valid) {
  const int i = blockIdx.x * blockDim.x + threadIdx.x;   // [0, B*N)
  if (i >= B_ * N_) return;
  const int b = i >> 8;
  const int n = i & 255;
  const int y = (n >> 4) << 5;
  const int x = (n & 15) << 5;
  const float v = vm[(size_t)b * HO_ * HO_ + (size_t)y * HO_ + x];
  valid[i] = (v > 0.5f) ? 1 : 0;
}

// ---------------------------------------------------------------------------
// Kernel 3: per-batch 256x256 similarity GEMM (K=256) via WMMA f16->f32.
// One wave per 16x16 output tile; 8 k-steps of V_WMMA_F32_16X16X32_F16.
// A/B fragments use contiguous k order (k = 16*hi + s), same in both (wave32).
// ---------------------------------------------------------------------------
__global__ void gemm_tiles(const _Float16* __restrict__ dA,   // depth [b][n][c]
                           const _Float16* __restrict__ dB,   // rgb   [b][m][c]
                           float* __restrict__ sim) {
  const int wid  = blockIdx.x * (blockDim.x >> 5) + (threadIdx.x >> 5); // [0, B*256)
  const int b    = wid >> 8;
  const int tile = wid & 255;
  const int n0   = (tile >> 4) << 4;
  const int m0   = (tile & 15) << 4;
  const int lane = threadIdx.x & 31;
  const int hi   = lane >> 4;     // half-wave selector
  const int ls   = lane & 15;

  const _Float16* Abase = dA + (((size_t)b << 8) + (n0 + ls)) * C_;
  const _Float16* Bbase = dB + (((size_t)b << 8) + (m0 + ls)) * C_;

  v8f acc = {};
#pragma unroll
  for (int kb = 0; kb < 8; ++kb) {
    // A 16x32 f16: row M = ls; lanes 0-15 hold K 0..15, lanes 16-31 K 16..31
    const int kA = kb * 32 + hi * 16;
    v16h af = *(const v16h*)(Abase + kA);
    // B 32x16 f16: column N = ls; lanes 0-15 hold K 0..15, lanes 16-31 K 16..31
    const int kB = kb * 32 + hi * 16;
    v16h bf = *(const v16h*)(Bbase + kB);
    acc = __builtin_amdgcn_wmma_f32_16x16x32_f16(
        /*neg_a=*/false, af, /*neg_b=*/false, bf,
        /*c_mod=*/(short)0, acc, /*reuse_a=*/false, /*reuse_b=*/false);
  }

  // C/D 16x16 f32: VGPR r -> row M = r + 8*hi, col N = ls
  float* out = sim + (((size_t)b << 8) + (n0 + hi * 8)) * N_ + (m0 + ls);
#pragma unroll
  for (int r = 0; r < 8; ++r)
    out[(size_t)r * N_] = acc[r];
}

// ---------------------------------------------------------------------------
// Kernel 4: per-row masked logsumexp + argmax + diag; block reduce per batch.
// One block per batch, thread t == row n.
// ---------------------------------------------------------------------------
__global__ void row_reduce(const float* __restrict__ sim, const int* __restrict__ valid,
                           float* __restrict__ bLoss, int* __restrict__ bStats) {
  const int b = blockIdx.x;
  const int t = threadIdx.x;             // row index 0..255
  __shared__ float vLDS[N_];
  __shared__ float redf[N_];
  __shared__ int   redi[N_];

  vLDS[t] = (float)valid[b * N_ + t];
  __syncthreads();

  const float* row = sim + (((size_t)b << 8) + t) * N_;
  float mx = -__builtin_inff();
  float sum = 0.f;
  float bestv = -__builtin_inff();
  int   besti = 0;
  for (int m = 0; m < N_; ++m) {
    const float s  = row[m] * INV_T;
    const float sm = (vLDS[m] != 0.f) ? s : -1e30f;
    if (sm > mx) { sum = sum * __expf(mx - sm) + 1.0f; mx = sm; }
    else         { sum += __expf(sm - mx); }
    if (sm > bestv) { bestv = sm; besti = m; }   // strict > => first max (argmax)
  }
  const float logZ = mx + __logf(sum);
  const float diag = row[t] * INV_T;
  const bool  v    = (vLDS[t] != 0.f);

  redf[t] = v ? (logZ - diag) : 0.f;
  redi[t] = (v ? 1 : 0) | (((v && besti == t) ? 1 : 0) << 16);
  __syncthreads();
  for (int s = 128; s > 0; s >>= 1) {
    if (t < s) { redf[t] += redf[t + s]; redi[t] += redi[t + s]; }
    __syncthreads();
  }
  if (t == 0) {
    bLoss[b]  = redf[0];
    bStats[b] = redi[0];   // low 16: n_valid, high 16: correct
  }
}

// ---------------------------------------------------------------------------
// Kernel 5: final scalar reduction over 32 batches (one wave).
// ---------------------------------------------------------------------------
__global__ void finalize(const float* __restrict__ bLoss, const int* __restrict__ bStats,
                         float* __restrict__ out) {
  const int b = threadIdx.x;             // 0..31
  const int st = bStats[b];
  const int nv = st & 0xFFFF;
  const int cr = (st >> 16) & 0xFFFF;
  const bool include = (nv >= 2);
  float lcontrib = include ? (bLoss[b] / (float)max(nv, 1)) : 0.f;
  int   tc = include ? nv : 0;
  int   cc = include ? cr : 0;
#pragma unroll
  for (int m = 16; m > 0; m >>= 1) {
    lcontrib += __shfl_xor(lcontrib, m, 32);
    tc       += __shfl_xor(tc, m, 32);
    cc       += __shfl_xor(cc, m, 32);
  }
  if (b == 0) {
    float avg = lcontrib / (float)B_;
    float acc = 0.f;
    if (tc > 0) acc = (float)cc / (float)max(tc, 1) * 100.0f;
    else        avg = 0.f;
    out[0] = avg;
    out[1] = acc;
  }
}

// ---------------------------------------------------------------------------
extern "C" void kernel_launch(void* const* d_in, const int* in_sizes, int n_in,
                              void* d_out, int out_size, void* d_ws, size_t ws_size,
                              hipStream_t stream) {
  const float* fmap_rgb   = (const float*)d_in[0];
  const float* fmap_depth = (const float*)d_in[1];
  // d_in[2] = projected_coords: provably unused by the outputs (bilinear result dead)
  const float* valid_mask = (const float*)d_in[3];
  float* out = (float*)d_out;

  char* ws = (char*)d_ws;
  _Float16* dDepth = (_Float16*)(ws);                          // 4 MB
  _Float16* dRgb   = (_Float16*)(ws + (4u  << 20));            // 4 MB
  float*    dSim   = (float*)   (ws + (8u  << 20));            // 8 MB
  int*      dValid = (int*)     (ws + (16u << 20));            // 32 KB
  float*    dBLoss = (float*)   (ws + (16u << 20) + (32u << 10));
  int*      dBStat = (int*)     ((char*)dBLoss + B_ * sizeof(float));

  // 2*B*N = 16384 waves, 8 waves/block -> 2048 blocks
  gather_norm<<<2048, 256, 0, stream>>>(fmap_depth, fmap_rgb, dDepth, dRgb);
  compute_valid<<<(B_ * N_ + 255) / 256, 256, 0, stream>>>(valid_mask, dValid);
  // B*256 tiles = 8192 waves, 8 waves/block -> 1024 blocks
  gemm_tiles<<<1024, 256, 0, stream>>>(dDepth, dRgb, dSim);
  row_reduce<<<B_, N_, 0, stream>>>(dSim, dValid, dBLoss, dBStat);
  finalize<<<1, 32, 0, stream>>>(dBLoss, dBStat, out);
}
